// ForwardEulerSimulator_47957604827830
// MI455X (gfx1250) — matrix operand order repaired and checked by reading, not compiled, not run on hardware
//
#include <hip/hip_runtime.h>

// ---------------- problem constants ----------------
#define MSTEPS 512
#define QB     8192
#define NX     8
#define NU     4
#define NUI    4
#define HD     512

#define WAVES_PER_BLOCK 8
#define TPB  (WAVES_PER_BLOCK * 32)
#define SPB  (WAVES_PER_BLOCK * 16)   // samples per block (16 per wave)

// LDS weight-tile blob offsets (in bf16 elements)
// A1 tiles are stored as (self,send) pairs per 32-wide hidden chunk.
#define OFF_A1P 0        // 32 tiles * 512 : W1_pop^T (+b1 row at k=12), A-layout, S/X order
#define OFF_A1I 16384    // 32 tiles * 512 : W1_ind^T (+b1 row at k=16), S/X order
#define OFF_A2P 32768    // 16 chunks * 512: W2_pop^T rows 0..7 (pad 16), K permuted by pi
#define OFF_A2I 40960    // 16 chunks * 512: W2_ind col0 as row 0 (pad), K permuted by pi
#define LDS_ELEMS 49152  // 96 KB

typedef __attribute__((ext_vector_type(16))) __bf16 v16bf;
typedef __attribute__((ext_vector_type(8)))  __bf16 v8bf;
typedef __attribute__((ext_vector_type(8)))  float  v8f;

union BfOp {
  v16bf v;
  v8bf  half8[2];
  __bf16 b[16];
  unsigned int u32[8];
};

// pack two f32 -> two bf16 in one dword: single v_cvt_pk_bf16_f32 (no builtin
// exposed for the 2-source form, so emit it directly)
static __device__ __forceinline__ unsigned int pkbf(float a, float b) {
  unsigned int r;
  asm("v_cvt_pk_bf16_f32 %0, %1, %2" : "=v"(r) : "v"(a), "v"(b));
  return r;
}

// half-wave swap: lane l <-> lane l^16 (SWAPX16 via ds_swizzle), on packed data
static __device__ __forceinline__ unsigned int swzu(unsigned int x) {
#if __has_builtin(__builtin_amdgcn_ds_swizzle)
  return (unsigned int)__builtin_amdgcn_ds_swizzle((int)x, 0x401F);
#else
  return (unsigned int)__shfl_xor((int)x, 16, 32);
#endif
}

static __device__ __forceinline__ float fast_tanh(float x) {
#if __has_builtin(__builtin_amdgcn_tanhf)
  return __builtin_amdgcn_tanhf(x);   // v_tanh_f32 (confirmed in disasm)
#else
  float xc = fminf(fmaxf(x, -10.0f), 10.0f);
  float t  = __builtin_amdgcn_exp2f(xc * 2.885390081777927f);
  return (t - 1.0f) * __builtin_amdgcn_rcpf(t + 1.0f);
#endif
}

static __device__ __forceinline__ v8f vzero8() {
  v8f z;
#pragma unroll
  for (int r = 0; r < 8; ++r) z[r] = 0.0f;
  return z;
}

__global__ void __launch_bounds__(TPB, 1)
fes_kernel(const float* __restrict__ x0,
           const float* __restrict__ u_all,
           const float* __restrict__ ui_all,
           const float* __restrict__ W1p, const float* __restrict__ b1p,
           const float* __restrict__ W2p, const float* __restrict__ b2p,
           const float* __restrict__ W1i, const float* __restrict__ b1i,
           const float* __restrict__ W2i, const float* __restrict__ b2i,
           float* __restrict__ out)
{
  extern __shared__ __bf16 smem[];

  const int tid = threadIdx.x;

  // -------- one-time per-block weight prep: bf16 tiles in WMMA A-operand order --------
  // A tile order used here (16x32, contiguous k, same as B1/B2): lane l, halves e=0..15:
//   m = l%16, g = l/16, k = 16g + e
  //
  // Layer-1 tiles per 32-hidden chunk j come in (self, send) pairs:
  //   self: rows 0..7 -> chunk rows 0..7,   rows 8..15 -> chunk rows 24..31
  //   send: rows 0..7 -> chunk rows 16..23, rows 8..15 -> chunk rows 8..15
  for (int d = tid; d < 16384; d += TPB) {
    int t = d >> 9, rem = d & 511, l = rem >> 4, e = rem & 15;
    int m = l & 15, g = l >> 4;
    int k = 16 * g + e;
    int j = t >> 1, isX = t & 1;
    int hid = 32 * j + m + (((((m >> 3) ^ isX) & 1) != 0) ? 16 : 0);
    float vp = (k < 12) ? W1p[k * HD + hid] : ((k == 12) ? b1p[hid] : 0.0f);
    float vi = (k < 16) ? W1i[k * HD + hid] : ((k == 16) ? b1i[hid] : 0.0f);
    smem[OFF_A1P + d] = (__bf16)vp;
    smem[OFF_A1I + d] = (__bf16)vi;
  }
  // Layer-2 tiles: K permuted by pi(k) = k<16 ? k : k^8, matching the B2 operand
  // assembled as [self regs | swizzled send regs] with no lane selects.
  for (int d = tid; d < 8192; d += TPB) {
    int c = d >> 9, rem = d & 511, l = rem >> 4, e = rem & 15;
    int m = l & 15, g = l >> 4;
    int k = 16 * g + e;
    int kp = (k < 16) ? k : (k ^ 8);
    int hid = 32 * c + kp;
    float vp = (m < 8) ? W2p[hid * NX + m] : 0.0f;   // A2p row m = channel m
    float vi = (m == 0) ? W2i[hid * NX] : 0.0f;      // A2i row 0 = ind channel 0
    smem[OFF_A2P + d] = (__bf16)vp;
    smem[OFF_A2I + d] = (__bf16)vi;
  }
  __syncthreads();

  const int lane = tid & 31;
  const int g    = lane >> 4;
  const int n    = lane & 15;
  const int wave = tid >> 5;
  const int s    = blockIdx.x * SPB + wave * 16 + n;   // global sample

  auto ldA = [&](int off) -> v16bf {
    BfOp t;
    t.half8[0] = *(const v8bf*)(smem + off + lane * 16);
    t.half8[1] = *(const v8bf*)(smem + off + lane * 16 + 8);
    return t.v;
  };

  // layer-2 bias (accumulator init); only g==0 rows 0..7 are real channels
  float b2reg[8];
#pragma unroll
  for (int r = 0; r < 8; ++r) b2reg[r] = (g == 0) ? b2p[r] : 0.0f;
  const float b2i0 = (g == 0) ? b2i[0] : 0.0f;

  // state x, replicated across half-waves (only g==0 copy is consumed)
  float x[8];
  {
    const float4* p = (const float4*)(x0 + (size_t)s * NX);
    float4 a = p[0], b = p[1];
    x[0] = a.x; x[1] = a.y; x[2] = a.z; x[3] = a.w;
    x[4] = b.x; x[5] = b.y; x[6] = b.z; x[7] = b.w;
  }

  for (int step = 0; step < MSTEPS; ++step) {
    // X_sim stores the PRE-update state
    if (lane < 16) {
      float4* po = (float4*)(out + (size_t)step * QB * NX + (size_t)s * NX);
      po[0] = make_float4(x[0], x[1], x[2], x[3]);
      po[1] = make_float4(x[4], x[5], x[6], x[7]);
    }

    float4 uv  = *(const float4*)(u_all  + ((size_t)step * QB + s) * NU);
    float4 uiv = *(const float4*)(ui_all + ((size_t)step * QB + s) * NUI);

    // B1 operand (K=input x N=sample). lanes g=0 hold k=0..15, g=1 hold k=16..31.
    // bf16(1.0) = 0x3F80 for the bias rows.
    BfOp B1pop, B1ind;
    if (g == 0) {
      unsigned int x01 = pkbf(x[0], x[1]), x23 = pkbf(x[2], x[3]);
      unsigned int x45 = pkbf(x[4], x[5]), x67 = pkbf(x[6], x[7]);
      unsigned int u01 = pkbf(uv.x, uv.y), u23 = pkbf(uv.z, uv.w);
      B1pop.u32[0] = x01; B1pop.u32[1] = x23; B1pop.u32[2] = x45; B1pop.u32[3] = x67;
      B1pop.u32[4] = u01; B1pop.u32[5] = u23;
      B1pop.u32[6] = 0x00003F80u;                  // bias row k=12 (pop)
      B1pop.u32[7] = 0u;
      B1ind.u32[0] = x01; B1ind.u32[1] = x23; B1ind.u32[2] = x45; B1ind.u32[3] = x67;
      B1ind.u32[4] = u01; B1ind.u32[5] = u23;
      B1ind.u32[6] = pkbf(uiv.x, uiv.y);
      B1ind.u32[7] = pkbf(uiv.z, uiv.w);
    } else {
#pragma unroll
      for (int q = 0; q < 8; ++q) { B1pop.u32[q] = 0u; B1ind.u32[q] = 0u; }
      B1ind.u32[0] = 0x00003F80u;                  // bias row k=16 (ind)
    }

    v8f accP = vzero8(), accI = vzero8();
#pragma unroll
    for (int r = 0; r < 8; ++r) accP[r] = b2reg[r];
    accI[0] = b2i0;

#pragma unroll 4
    for (int j = 0; j < 16; ++j) {
      // ---- layer 1: self + send tiles for both nets (4 WMMAs) ----
      v8f zsP = __builtin_amdgcn_wmma_f32_16x16x32_bf16(
          false, ldA(OFF_A1P + (2 * j + 0) * 512), false, B1pop.v, (short)0, vzero8(), false, false);
      v8f zxP = __builtin_amdgcn_wmma_f32_16x16x32_bf16(
          false, ldA(OFF_A1P + (2 * j + 1) * 512), false, B1pop.v, (short)0, vzero8(), false, false);
      v8f zsI = __builtin_amdgcn_wmma_f32_16x16x32_bf16(
          false, ldA(OFF_A1I + (2 * j + 0) * 512), false, B1ind.v, (short)0, vzero8(), false, false);
      v8f zxI = __builtin_amdgcn_wmma_f32_16x16x32_bf16(
          false, ldA(OFF_A1I + (2 * j + 1) * 512), false, B1ind.v, (short)0, vzero8(), false, false);

      // ---- tanh, pack bf16 pairs; low halves = self regs, high halves = swizzled send regs.
      // No lane-dependent selects: the permutation lives in the A1 tile order + A2 pi(k).
      BfOp B2p, B2i;
#pragma unroll
      for (int q = 0; q < 4; ++q) {
        B2p.u32[q]     = pkbf(fast_tanh(zsP[2 * q]), fast_tanh(zsP[2 * q + 1]));
        B2p.u32[4 + q] = swzu(pkbf(fast_tanh(zxP[2 * q]), fast_tanh(zxP[2 * q + 1])));
        B2i.u32[q]     = pkbf(fast_tanh(zsI[2 * q]), fast_tanh(zsI[2 * q + 1]));
        B2i.u32[4 + q] = swzu(pkbf(fast_tanh(zxI[2 * q]), fast_tanh(zxI[2 * q + 1])));
      }

      // ---- layer 2: dx^T accumulate (2 WMMAs) ----
      accP = __builtin_amdgcn_wmma_f32_16x16x32_bf16(
          false, ldA(OFF_A2P + j * 512), false, B2p.v, (short)0, accP, false, false);
      accI = __builtin_amdgcn_wmma_f32_16x16x32_bf16(
          false, ldA(OFF_A2I + j * 512), false, B2i.v, (short)0, accI, false, false);
    }

    // ---- Euler update (TS = 1.0), clamp CGM channel 0 ----
#pragma unroll
    for (int r = 0; r < 8; ++r) x[r] = x[r] + accP[r];
    x[0] += accI[0];
    x[0] = fminf(fmaxf(x[0], -2.0f), 3.0f);
  }
}

extern "C" void kernel_launch(void* const* d_in, const int* in_sizes, int n_in,
                              void* d_out, int out_size, void* d_ws, size_t ws_size,
                              hipStream_t stream) {
  const float* x0  = (const float*)d_in[0];
  const float* u   = (const float*)d_in[1];
  const float* ui  = (const float*)d_in[2];
  const float* W1p = (const float*)d_in[3];
  const float* b1p = (const float*)d_in[4];
  const float* W2p = (const float*)d_in[5];
  const float* b2p = (const float*)d_in[6];
  const float* W1i = (const float*)d_in[7];
  const float* b1i = (const float*)d_in[8];
  const float* W2i = (const float*)d_in[9];
  const float* b2i = (const float*)d_in[10];
  float* out = (float*)d_out;

  (void)in_sizes; (void)n_in; (void)out_size; (void)d_ws; (void)ws_size;

  hipFuncSetAttribute((const void*)fes_kernel,
                      hipFuncAttributeMaxDynamicSharedMemorySize,
                      LDS_ELEMS * (int)sizeof(__bf16));

  dim3 grid(QB / SPB);   // 64 blocks
  dim3 block(TPB);       // 256 threads = 8 waves
  fes_kernel<<<grid, block, LDS_ELEMS * sizeof(__bf16), stream>>>(
      x0, u, ui, W1p, b1p, W2p, b2p, W1i, b1i, W2i, b2i, out);
}
